// OptimizedSwinMLPBlock_66924180406728
// MI455X (gfx1250) — compile-verified, matrix operand order checked by reading, not run on hardware
//
#include <hip/hip_runtime.h>
#include <hip/hip_bf16.h>

typedef __bf16 bf16;
typedef __attribute__((ext_vector_type(16))) __bf16 v16bf;
typedef __attribute__((ext_vector_type(8)))  __bf16 v8bf;
typedef __attribute__((ext_vector_type(8)))  float  v8f;
typedef __attribute__((ext_vector_type(4)))  int    v4i;

#define AS1 __attribute__((address_space(1)))
#define AS3 __attribute__((address_space(3)))

// gfx1250 async global->LDS DMA path (ASYNCcnt-tracked), guarded so the
// source compiles on toolchains that don't expose the builtins.
#if defined(__has_builtin)
#if __has_builtin(__builtin_amdgcn_global_load_async_to_lds_b128) && \
    __has_builtin(__builtin_amdgcn_s_wait_asynccnt)
#define HAVE_ASYNC_LDS 1
#endif
#endif

// Problem constants
constexpr int BB   = 32;
constexpr int HH   = 56;
constexpr int WW   = 56;
constexpr int CC   = 256;
constexpr int NHD  = 8;
constexpr int WS   = 7;
constexpr int SS   = 49;      // WS*WS
constexpr int PAD  = 4;       // WS - SHIFT
constexpr int NWIN = 9;       // 63/7 windows per axis
constexpr int TOK  = BB * HH * WW;   // 100352
constexpr int HID  = 4 * CC;         // 1024

// ---------------------------------------------------------------------------
// WMMA helpers. CDNA5 16-bit A/B fragment layout (CDNA5 ISA §7.12.2):
// lane L, element e -> k = (e>>3)*16 + (L>>4)*8 + (e&7).
// ---------------------------------------------------------------------------
__device__ __forceinline__ v8f wmma_bf16(v16bf a, v16bf b, v8f c) {
  return __builtin_amdgcn_wmma_f32_16x16x32_bf16(
      false, a, false, b, (short)0, c, false, false);
}

__device__ __forceinline__ v16bf frag2x8(const bf16* base) {
  union { uint4 u; v8bf b; } lo, hi;
  lo.u = *(const uint4*)(base);
  hi.u = *(const uint4*)(base + 16);
  return __builtin_shufflevector(lo.b, hi.b,
                                 0, 1, 2, 3, 4, 5, 6, 7,
                                 8, 9, 10, 11, 12, 13, 14, 15);
}

// Stage one 16-byte chunk global -> LDS: async DMA if available.
__device__ __forceinline__ void stage_b128(const bf16* g, bf16* l) {
#ifdef HAVE_ASYNC_LDS
  __builtin_amdgcn_global_load_async_to_lds_b128(
      (AS1 v4i*)g, (AS3 v4i*)l, 0, 0);
#else
  *(uint4*)l = *(const uint4*)g;
#endif
}

__device__ __forceinline__ void stage_wait() {
#ifdef HAVE_ASYNC_LDS
  __builtin_amdgcn_s_wait_asynccnt(0);
#endif
}

// ---------------------------------------------------------------------------
// Weight packing: B-matrix [K][N] fp32 -> fragment-ready bf16.
// tile_id = n_sub*(K/32)+k_t, then [lane][e] contiguous (32 B per lane).
// ---------------------------------------------------------------------------
__global__ void pack_b_kernel(const float* __restrict__ w,
                              bf16* __restrict__ out, int K, int N) {
  int idx = blockIdx.x * blockDim.x + threadIdx.x;
  if (idx >= K * N) return;
  int tile = idx >> 9;
  int r    = idx & 511;
  int L    = r >> 4;
  int e    = r & 15;
  int kt   = K >> 5;
  int n_sub = tile / kt;
  int k_t   = tile - n_sub * kt;
  int n = (n_sub << 4) + (L & 15);
  int k = (k_t << 5) + ((L >> 4) << 3) + ((e >> 3) << 4) + (e & 7);
  out[idx] = (bf16)w[k * N + n];
}

// Spatial weights: per head, 49x49 zero-padded to 64x64, tiles (kt, mt) of
// 16x32, fragment-ready. tile_id = ((head*2 + kt)*4 + mt).
__global__ void pack_spatial_kernel(const float* __restrict__ sw,
                                    bf16* __restrict__ out) {
  int idx = blockIdx.x * blockDim.x + threadIdx.x;
  if (idx >= NHD * 8 * 512) return;
  int tile = idx >> 9;
  int r    = idx & 511;
  int L    = r >> 4;
  int e    = r & 15;
  int mt   = tile & 3;
  int kt   = (tile >> 2) & 1;
  int h    = tile >> 3;
  int m = (mt << 4) + (L & 15);
  int k = (kt << 5) + ((L >> 4) << 3) + ((e >> 3) << 4) + (e & 7);
  float v = (m < SS && k < SS) ? sw[(h * SS + m) * SS + k] : 0.0f;
  out[idx] = (bf16)v;
}

// ---------------------------------------------------------------------------
// LayerNorm: one wave32 per token. Lane owns 8 CONTIGUOUS channels so all
// traffic is b128: 2x float4 in, 2x2 float4 (w,b), 1x uint4 bf16 out.
// ---------------------------------------------------------------------------
__global__ void ln_kernel(const float* __restrict__ x,
                          const float* __restrict__ w,
                          const float* __restrict__ b,
                          bf16* __restrict__ out, int ntok) {
  int wave = (int)((blockIdx.x * blockDim.x + threadIdx.x) >> 5);
  int lane = threadIdx.x & 31;
  if (wave >= ntok) return;
  const float4* xr = (const float4*)(x + (size_t)wave * CC + lane * 8);
  float4 lo = xr[0], hi = xr[1];
  float v[8] = {lo.x, lo.y, lo.z, lo.w, hi.x, hi.y, hi.z, hi.w};
  float s = 0.f, s2 = 0.f;
#pragma unroll
  for (int i = 0; i < 8; ++i) { s += v[i]; s2 += v[i] * v[i]; }
#pragma unroll
  for (int off = 16; off > 0; off >>= 1) {
    s  += __shfl_xor(s,  off, 32);
    s2 += __shfl_xor(s2, off, 32);
  }
  float mu  = s * (1.0f / CC);
  float var = s2 * (1.0f / CC) - mu * mu;
  float inv = rsqrtf(var + 1e-5f);
  const float4* wr = (const float4*)(w + lane * 8);
  const float4* br = (const float4*)(b + lane * 8);
  float4 w0 = wr[0], w1 = wr[1], b0 = br[0], b1 = br[1];
  float wv[8] = {w0.x, w0.y, w0.z, w0.w, w1.x, w1.y, w1.z, w1.w};
  float bv[8] = {b0.x, b0.y, b0.z, b0.w, b1.x, b1.y, b1.z, b1.w};
  union { uint4 u; v8bf bb; } pk;
#pragma unroll
  for (int i = 0; i < 8; ++i)
    pk.bb[i] = (bf16)((v[i] - mu) * inv * wv[i] + bv[i]);
  *(uint4*)(out + (size_t)wave * CC + lane * 8) = pk.u;
}

// ---------------------------------------------------------------------------
// Spatial MLP: one block per shifted window (2592 blocks, 256 thr = 8 waves).
// Window staged TRANSPOSED into LDS: lwin_t[ch][j] (256 x 64 bf16, 32 KB).
// Wave w computes head w: out(49x32) = wgt(49x49) x win(49x32), 16 WMMAs.
// ---------------------------------------------------------------------------
__global__ void spatial_kernel(const float* __restrict__ x,
                               const bf16*  __restrict__ xln,
                               const bf16*  __restrict__ swp,
                               const float* __restrict__ sb,
                               float* __restrict__ x1) {
  __shared__ __align__(16) bf16 lwin_t[256 * 64];

  int wi  = blockIdx.x;
  int bat = wi / (NWIN * NWIN);
  int rem = wi - bat * (NWIN * NWIN);
  int wy  = rem / NWIN;
  int wx  = rem - wy * NWIN;

  // Stage window transposed: thread = channel, 8 j's packed per b128 store.
  {
    int ch = threadIdx.x;
    const bf16* xb = xln + (size_t)bat * HH * WW * CC + ch;
#pragma unroll
    for (int jc = 0; jc < 8; ++jc) {
      union { uint4 u; v8bf b; } pk;
#pragma unroll
      for (int t = 0; t < 8; ++t) {
        int j = jc * 8 + t;
        bf16 val = (bf16)0.0f;
        if (j < SS) {
          int r = j / WS, c = j - (j / WS) * WS;
          int oh = wy * WS + r - PAD;
          int ow = wx * WS + c - PAD;
          if (oh >= 0 && oh < HH && ow >= 0 && ow < WW)
            val = xb[((size_t)oh * WW + ow) * CC];
        }
        pk.b[t] = val;
      }
      *(uint4*)(lwin_t + ch * 64 + jc * 8) = pk.u;
    }
  }
  __syncthreads();

  int wave = threadIdx.x >> 5;
  int lane = threadIdx.x & 31;
  int hd   = wave;                      // 8 waves <-> 8 heads
  int half = lane >> 4;
  int n    = lane & 15;

  v8f acc[4][2];
#pragma unroll
  for (int mt = 0; mt < 4; ++mt)
#pragma unroll
    for (int nt = 0; nt < 2; ++nt) acc[mt][nt] = (v8f){};

#pragma unroll
  for (int kt = 0; kt < 2; ++kt) {
    v16bf af[4], bfrag[2];
#pragma unroll
    for (int mt = 0; mt < 4; ++mt)   // pre-packed: contiguous v16bf per lane
      af[mt] = *(const v16bf*)(swp + (((hd * 2 + kt) * 4 + mt) << 9) + lane * 16);
#pragma unroll
    for (int nt = 0; nt < 2; ++nt) {
      int ch = hd * 32 + nt * 16 + n;
      bfrag[nt] = frag2x8(lwin_t + ch * 64 + kt * 32 + half * 8);
    }
#pragma unroll
    for (int mt = 0; mt < 4; ++mt)
#pragma unroll
      for (int nt = 0; nt < 2; ++nt)
        acc[mt][nt] = wmma_bf16(af[mt], bfrag[nt], acc[mt][nt]);
  }

  // Epilogue: token/validity computed once per output row, then 2 stores.
#pragma unroll
  for (int mt = 0; mt < 4; ++mt) {
#pragma unroll
    for (int v = 0; v < 8; ++v) {
      int m = mt * 16 + half * 8 + v;   // window position index i
      if (m < SS) {
        int r = m / WS, c = m - (m / WS) * WS;
        int oh = wy * WS + r - PAD;
        int ow = wx * WS + c - PAD;
        if (oh >= 0 && oh < HH && ow >= 0 && ow < WW) {
          size_t base = ((size_t)bat * HH * WW + (size_t)oh * WW + ow) * CC;
          float sbv = sb[hd * SS + m];
#pragma unroll
          for (int nt = 0; nt < 2; ++nt) {
            int ch = hd * 32 + nt * 16 + n;
            x1[base + ch] = x[base + ch] + acc[mt][nt][v] + sbv;
          }
        }
      }
    }
  }
}

// ---------------------------------------------------------------------------
// GEMM fc1: [100352x256]bf16 @ [256x1024]bf16(packed) + bias -> GELU -> bf16
// Block tile 64x128, 8 waves each own 32x32 (2x2 frags), K-step 32.
// A staged into LDS via async DMA (ASYNCcnt) when available; B fragments
// direct from packed global (L2-hot, 32 B contiguous per lane).
// ---------------------------------------------------------------------------
__global__ void gemm_fc1(const bf16* __restrict__ Ag,
                         const bf16* __restrict__ Bp,
                         const float* __restrict__ bias,
                         bf16* __restrict__ out) {
  __shared__ __align__(16) bf16 lA[64 * 32];
  const int KT = CC / 32;                   // 8 k-tiles
  const int NT = HID / 128;                 // 8 n-blocks
  int m0  = (blockIdx.x / NT) * 64;
  int n0  = (blockIdx.x % NT) * 128;
  int tid = threadIdx.x;
  int wave = tid >> 5, lane = tid & 31;
  int wm = wave & 1, wn = wave >> 1;
  int half = lane >> 4;

  v8f acc[2][2];
#pragma unroll
  for (int mt = 0; mt < 2; ++mt)
#pragma unroll
    for (int nt = 0; nt < 2; ++nt) acc[mt][nt] = (v8f){};

  const bf16* bptr[2];
#pragma unroll
  for (int nt = 0; nt < 2; ++nt)
    bptr[nt] = Bp + (size_t)((n0 >> 4) + wn * 2 + nt) * KT * 512 + lane * 16;

  int ar = tid >> 2, ac = (tid & 3) * 8;    // A staging: 1 b128 per thread
  const bf16* asrc = Ag + (size_t)(m0 + ar) * CC + ac;
  bf16* adst = lA + ar * 32 + ac;

  for (int k0 = 0; k0 < CC; k0 += 32) {
    stage_b128(asrc + k0, adst);
    if (k0 + 32 < CC) {
      __builtin_prefetch(asrc + k0 + 32, 0, 1);
      __builtin_prefetch(bptr[0] + 512, 0, 1);
    }
    stage_wait();
    __syncthreads();
    v16bf af[2], bfrag[2];
#pragma unroll
    for (int mt = 0; mt < 2; ++mt)
      af[mt] = frag2x8(lA + (wm * 32 + mt * 16 + (lane & 15)) * 32 + half * 8);
#pragma unroll
    for (int nt = 0; nt < 2; ++nt) {
      bfrag[nt] = *(const v16bf*)(bptr[nt]);
      bptr[nt] += 512;
    }
#pragma unroll
    for (int mt = 0; mt < 2; ++mt)
#pragma unroll
      for (int nt = 0; nt < 2; ++nt)
        acc[mt][nt] = wmma_bf16(af[mt], bfrag[nt], acc[mt][nt]);
    __syncthreads();
  }

  int n = lane & 15;
#pragma unroll
  for (int mt = 0; mt < 2; ++mt) {
#pragma unroll
    for (int nt = 0; nt < 2; ++nt) {
      int col = n0 + wn * 32 + nt * 16 + n;
      float bv = bias[col];
#pragma unroll
      for (int v = 0; v < 8; ++v) {
        int row = m0 + wm * 32 + mt * 16 + half * 8 + v;
        float val = acc[mt][nt][v] + bv;
        float g = 0.5f * val * (1.0f + erff(val * 0.70710678118654752f));
        out[(size_t)row * HID + col] = (bf16)g;
      }
    }
  }
}

// ---------------------------------------------------------------------------
// GEMM fc2: [100352x1024]bf16 @ [1024x256]bf16(packed) + bias + residual
// -> fp32 out. Same structure, K=1024.
// ---------------------------------------------------------------------------
__global__ void gemm_fc2(const bf16* __restrict__ Ag,
                         const bf16* __restrict__ Bp,
                         const float* __restrict__ bias,
                         const float* __restrict__ resid,
                         float* __restrict__ out) {
  __shared__ __align__(16) bf16 lA[64 * 32];
  const int KT = HID / 32;                  // 32 k-tiles
  const int NT = CC / 128;                  // 2 n-blocks
  int m0  = (blockIdx.x / NT) * 64;
  int n0  = (blockIdx.x % NT) * 128;
  int tid = threadIdx.x;
  int wave = tid >> 5, lane = tid & 31;
  int wm = wave & 1, wn = wave >> 1;
  int half = lane >> 4;

  v8f acc[2][2];
#pragma unroll
  for (int mt = 0; mt < 2; ++mt)
#pragma unroll
    for (int nt = 0; nt < 2; ++nt) acc[mt][nt] = (v8f){};

  const bf16* bptr[2];
#pragma unroll
  for (int nt = 0; nt < 2; ++nt)
    bptr[nt] = Bp + (size_t)((n0 >> 4) + wn * 2 + nt) * KT * 512 + lane * 16;

  int ar = tid >> 2, ac = (tid & 3) * 8;
  const bf16* asrc = Ag + (size_t)(m0 + ar) * HID + ac;
  bf16* adst = lA + ar * 32 + ac;

  for (int k0 = 0; k0 < HID; k0 += 32) {
    stage_b128(asrc + k0, adst);
    if (k0 + 32 < HID) {
      __builtin_prefetch(asrc + k0 + 32, 0, 1);
      __builtin_prefetch(bptr[0] + 512, 0, 1);
    }
    stage_wait();
    __syncthreads();
    v16bf af[2], bfrag[2];
#pragma unroll
    for (int mt = 0; mt < 2; ++mt)
      af[mt] = frag2x8(lA + (wm * 32 + mt * 16 + (lane & 15)) * 32 + half * 8);
#pragma unroll
    for (int nt = 0; nt < 2; ++nt) {
      bfrag[nt] = *(const v16bf*)(bptr[nt]);
      bptr[nt] += 512;
    }
#pragma unroll
    for (int mt = 0; mt < 2; ++mt)
#pragma unroll
      for (int nt = 0; nt < 2; ++nt)
        acc[mt][nt] = wmma_bf16(af[mt], bfrag[nt], acc[mt][nt]);
    __syncthreads();
  }

  int n = lane & 15;
#pragma unroll
  for (int mt = 0; mt < 2; ++mt) {
#pragma unroll
    for (int nt = 0; nt < 2; ++nt) {
      int col = n0 + wn * 32 + nt * 16 + n;
      float bv = bias[col];
#pragma unroll
      for (int v = 0; v < 8; ++v) {
        int row = m0 + wm * 32 + mt * 16 + half * 8 + v;
        size_t o = (size_t)row * CC + col;
        out[o] = acc[mt][nt][v] + bv + resid[o];
      }
    }
  }
}

// ---------------------------------------------------------------------------
extern "C" void kernel_launch(void* const* d_in, const int* in_sizes, int n_in,
                              void* d_out, int out_size, void* d_ws, size_t ws_size,
                              hipStream_t stream) {
  (void)in_sizes; (void)n_in; (void)out_size; (void)ws_size;

  const float* x    = (const float*)d_in[0];
  const float* n1w  = (const float*)d_in[1];
  const float* n1b  = (const float*)d_in[2];
  const float* sw   = (const float*)d_in[3];
  const float* sb   = (const float*)d_in[4];
  const float* n2w  = (const float*)d_in[5];
  const float* n2b  = (const float*)d_in[6];
  const float* fc1w = (const float*)d_in[7];
  const float* fc1b = (const float*)d_in[8];
  const float* fc2w = (const float*)d_in[9];
  const float* fc2b = (const float*)d_in[10];
  float* out = (float*)d_out;

  // Workspace layout
  char* p = (char*)d_ws;
  bf16*  fc1wb = (bf16*)p;  p += (size_t)CC * HID * 2;          // 512 KB packed
  bf16*  fc2wb = (bf16*)p;  p += (size_t)HID * CC * 2;          // 512 KB packed
  bf16*  swp   = (bf16*)p;  p += (size_t)NHD * 8 * 512 * 2;     // 64 KB packed
  bf16*  xln1  = (bf16*)p;  p += (size_t)TOK * CC * 2;          // 51 MB
  float* x1    = (float*)p; p += (size_t)TOK * CC * 4;          // 103 MB
  bf16*  xn    = (bf16*)p;  p += (size_t)TOK * CC * 2;          // 51 MB
  bf16*  hbuf  = (bf16*)p;  p += (size_t)TOK * HID * 2;         // 205 MB

  pack_b_kernel<<<(CC * HID + 255) / 256, 256, 0, stream>>>(fc1w, fc1wb, CC, HID);
  pack_b_kernel<<<(HID * CC + 255) / 256, 256, 0, stream>>>(fc2w, fc2wb, HID, CC);
  pack_spatial_kernel<<<(NHD * 8 * 512 + 255) / 256, 256, 0, stream>>>(sw, swp);
  ln_kernel<<<(TOK + 7) / 8, 256, 0, stream>>>(x, n1w, n1b, xln1, TOK);
  spatial_kernel<<<BB * NWIN * NWIN, 256, 0, stream>>>(x, xln1, swp, sb, x1);
  ln_kernel<<<(TOK + 7) / 8, 256, 0, stream>>>(x1, n2w, n2b, xn, TOK);
  gemm_fc1<<<(TOK / 64) * (HID / 128), 256, 0, stream>>>(xn, fc1wb, fc1b, hbuf);
  gemm_fc2<<<(TOK / 64) * (CC / 128), 256, 0, stream>>>(hbuf, fc2wb, fc2b, x1, out);
}
